// LPOLAttention_51874615001333
// MI455X (gfx1250) — matrix operand order repaired and checked by reading, not compiled, not run on hardware
//
#include <hip/hip_runtime.h>
#include <hip/hip_bf16.h>
#include <stdint.h>

typedef __attribute__((ext_vector_type(16))) __bf16 v16bf;
typedef __attribute__((ext_vector_type(8)))  float  v8f;
typedef unsigned short u16;

#define NUM_HEADS 16
#define EXP_DIM   256
#define HIDDEN    1024
#define BATCH     4
#define SEQ       2048
#define HEAD_DIM  64

union FragU {
    v16bf v;
    uint4 q[2];
    u16   u[16];
};

__device__ __forceinline__ u16 f2bf(float f) {
    uint32_t u = __float_as_uint(f);
    u += 0x7FFFu + ((u >> 16) & 1u);       // round-to-nearest-even
    return (u16)(u >> 16);
}
__device__ __forceinline__ float bf2f(u16 h) {
    return __uint_as_float(((uint32_t)h) << 16);
}

__device__ __forceinline__ v8f wmma_bf16(const FragU& a, const FragU& b, v8f c) {
    return __builtin_amdgcn_wmma_f32_16x16x32_bf16(
        /*neg_a=*/false, a.v, /*neg_b=*/false, b.v,
        /*c_mod=*/(short)0, c, /*reuse_a=*/false, /*reuse_b=*/false);
}

// Async DMA: 16 bytes global -> LDS, tracked by ASYNCcnt (no VGPR round-trip).
// LDS byte offset = low 32 bits of the generic pointer (aperture spec).
__device__ __forceinline__ void async_copy16(const u16* lds_dst, const u16* gsrc) {
    uint32_t loff = (uint32_t)(uintptr_t)lds_dst;
    asm volatile("global_load_async_to_lds_b128 %0, %1, off"
                 :: "v"(loff), "v"((uint64_t)(uintptr_t)gsrc)
                 : "memory");
}
#define ASYNC_WAIT_4() asm volatile("s_wait_asynccnt 0x4" ::: "memory")
#define ASYNC_WAIT_0() asm volatile("s_wait_asynccnt 0x0" ::: "memory")

// A fragment (16x32 bf16, ISA layout): lanes 0-15 hold row M=l16 with K 0..7 /
// 16..23, lanes 16-31 hold K 8..15 / 24..31. Two 16-byte chunks per lane.
__device__ __forceinline__ FragU loadA(const u16* p0, int stride, int kbase) {
    const int lane = threadIdx.x & 31;
    const int l16  = lane & 15;
    const int hi   = lane >> 4;
    const u16* p = p0 + (size_t)l16 * stride + kbase;
    FragU f;
    f.q[0] = *(const uint4*)(p + (hi ? 8  : 0));
    f.q[1] = *(const uint4*)(p + (hi ? 24 : 16));
    return f;
}

// B fragment (32x16 bf16): lane holds column N=l16; lanes 0-15 K 0..7 / 16..23,
// lanes 16-31 K 8..15 / 24..31 (two 16-byte chunks of a transposed-tile row).
__device__ __forceinline__ FragU loadB(const u16* p0, int stride, int kbase) {
    const int lane = threadIdx.x & 31;
    const int l16  = lane & 15;
    const int hi   = lane >> 4;
    const u16* p = p0 + (size_t)l16 * stride + kbase + (hi ? 8 : 0);
    FragU f;
    f.q[0] = *(const uint4*)(p);
    f.q[1] = *(const uint4*)(p + 16);
    return f;
}

// ---------------------------------------------------------------------------
// Elementwise fp32 -> bf16 convert (4 per thread).
// ---------------------------------------------------------------------------
__global__ __launch_bounds__(256) void convert_bf16_kernel(
    const float* __restrict__ s, u16* __restrict__ d, int n)
{
    int i = (blockIdx.x * 256 + threadIdx.x) * 4;
    if (i < n) {
        float4 f = *(const float4*)(s + i);
        d[i + 0] = f2bf(f.x); d[i + 1] = f2bf(f.y);
        d[i + 2] = f2bf(f.z); d[i + 3] = f2bf(f.w);
    }
}

// ---------------------------------------------------------------------------
// Wt[n][k] = bf16(W[k][n]) — 32x32 LDS tile transpose, 256 threads.
// ---------------------------------------------------------------------------
__global__ __launch_bounds__(256) void transpose_bf16_kernel(
    const float* __restrict__ W, u16* __restrict__ Wt, int K, int N)
{
    __shared__ float tile[32][33];
    const int k0 = blockIdx.x * 32, n0 = blockIdx.y * 32;
    const int tx = threadIdx.x & 31, ty = threadIdx.x >> 5;
    #pragma unroll
    for (int r = ty; r < 32; r += 8)
        tile[r][tx] = W[(size_t)(k0 + r) * N + n0 + tx];
    __syncthreads();
    #pragma unroll
    for (int r = ty; r < 32; r += 8)
        Wt[(size_t)(n0 + r) * K + k0 + tx] = f2bf(tile[tx][r]);
}

// ---------------------------------------------------------------------------
// Async double-buffered GEMM: C[M,N] = A[M,K] @ Bt[N,K]^T + bias[N]
// A, Bt are bf16 in global; tiles staged via global_load_async_to_lds_b128
// one K-step ahead (4 B128 DMAs/thread/tile), retired with s_wait_asynccnt.
// Block 128x128, K-step 32; 8 waves, each 64x32 (4x2 WMMA tiles).
// ---------------------------------------------------------------------------
template<typename OutT>
__global__ __launch_bounds__(256) void gemm_async_kernel(
    const u16* __restrict__ A, const u16* __restrict__ Bt,
    const float* __restrict__ bias, OutT* __restrict__ C,
    int M, int N, int K)
{
    __shared__ u16 As[2][128 * 40];
    __shared__ u16 Bs[2][128 * 40];

    const int tid  = threadIdx.x;
    const int wave = tid >> 5;
    const int lane = tid & 31;
    const int l16  = lane & 15;
    const int hi   = lane >> 4;
    const int m0   = blockIdx.y * 128;
    const int n0   = blockIdx.x * 128;
    const int wm   = wave & 1;     // 64-row group
    const int wn   = wave >> 1;    // 32-col group

    // staging role: 16 halfwords (= two B128 DMAs) of one row per tile
    const int sr = tid >> 1;             // row 0..127 (A: m, B: n)
    const int sc = (tid & 1) * 16;       // k segment 0 or 16

    v8f acc[4][2];
    #pragma unroll
    for (int i = 0; i < 4; ++i)
        #pragma unroll
        for (int j = 0; j < 2; ++j)
            #pragma unroll
            for (int r = 0; r < 8; ++r) acc[i][j][r] = 0.0f;

    const u16* gA = A  + (size_t)(m0 + sr) * K + sc;
    const u16* gB = Bt + (size_t)(n0 + sr) * K + sc;
    const int nT = K >> 5;

    // prologue: DMA tile 0 into buffer 0
    async_copy16(&As[0][sr * 40 + sc],     gA);
    async_copy16(&As[0][sr * 40 + sc + 8], gA + 8);
    async_copy16(&Bs[0][sr * 40 + sc],     gB);
    async_copy16(&Bs[0][sr * 40 + sc + 8], gB + 8);

    for (int t = 0; t < nT; ++t) {
        const int cur = t & 1;
        if (t + 1 < nT) {               // DMA next tile into other buffer
            const int k1 = (t + 1) << 5;
            async_copy16(&As[cur ^ 1][sr * 40 + sc],     gA + k1);
            async_copy16(&As[cur ^ 1][sr * 40 + sc + 8], gA + k1 + 8);
            async_copy16(&Bs[cur ^ 1][sr * 40 + sc],     gB + k1);
            async_copy16(&Bs[cur ^ 1][sr * 40 + sc + 8], gB + k1 + 8);
            ASYNC_WAIT_4();             // tile t's 4 DMAs retired (in-order)
        } else {
            ASYNC_WAIT_0();
        }
        __syncthreads();                // tile t visible to all waves

        FragU bfrag[2];
        #pragma unroll
        for (int j = 0; j < 2; ++j)
            bfrag[j] = loadB(&Bs[cur][(wn * 32 + j * 16) * 40], 40, 0);
        #pragma unroll
        for (int i = 0; i < 4; ++i) {
            FragU afrag = loadA(&As[cur][(wm * 64 + i * 16) * 40], 40, 0);
            #pragma unroll
            for (int j = 0; j < 2; ++j)
                acc[i][j] = wmma_bf16(afrag, bfrag[j], acc[i][j]);
        }
        __syncthreads();                // done reading buf before it's re-DMAed
    }

    // epilogue: bias + store (D layout: reg r -> row r / r+8)
    #pragma unroll
    for (int i = 0; i < 4; ++i) {
        const int rowb = m0 + wm * 64 + i * 16 + (hi ? 8 : 0);
        #pragma unroll
        for (int j = 0; j < 2; ++j) {
            const int col = n0 + wn * 32 + j * 16 + l16;
            const float bs = bias[col];
            #pragma unroll
            for (int r = 0; r < 8; ++r) {
                float v = acc[i][j][r] + bs;
                OutT* cp = C + (size_t)(rowb + r) * N + col;
                if constexpr (sizeof(OutT) == 4) *cp = v;
                else                             *cp = f2bf(v);
            }
        }
    }
}

// ---------------------------------------------------------------------------
// Flash attention with experience bias.
// Grid: (S/128, NUM_HEADS, B); 8 waves, each owning 16 q rows.
// scores = (Q*2^-3)Kᵀ + (EQ*cb)EKᵀ in one WMMA accumulator chain; online
// softmax; P transposed via per-wave LDS; V transposed via shared LDS tile.
// ---------------------------------------------------------------------------
__global__ __launch_bounds__(256) void flash_attn_kernel(
    const u16* __restrict__ Qb, const u16* __restrict__ Kb,
    const u16* __restrict__ Vb, const u16* __restrict__ EQb,
    const u16* __restrict__ EKb, const float* __restrict__ conf,
    u16* __restrict__ Ctxb)
{
    __shared__ u16 Vt[64 * 80];           // Vt[d][kv]
    __shared__ u16 Pw[8 * 16 * 80];       // per-wave P, A-layout source

    const int tid  = threadIdx.x;
    const int wave = tid >> 5;
    const int lane = tid & 31;
    const int l16  = lane & 15;
    const int hi   = lane >> 4;
    const int b    = blockIdx.z;
    const int h    = blockIdx.y;
    const int qw   = blockIdx.x * 128 + wave * 16;
    const size_t rowQ = (size_t)b * SEQ + qw;

    const float qscale = 0.125f;                  // 1/sqrt(64)
    const float cb     = 0.3f * conf[b] / 16.0f;  // 0.3 * conf / sqrt(256)

    FragU qf[2], eqf[8];
    #pragma unroll
    for (int ks = 0; ks < 2; ++ks) {
        qf[ks] = loadA(Qb + rowQ * HIDDEN + h * HEAD_DIM, HIDDEN, ks * 32);
        #pragma unroll
        for (int e = 0; e < 16; ++e) qf[ks].u[e] = f2bf(bf2f(qf[ks].u[e]) * qscale);
    }
    #pragma unroll
    for (int ks = 0; ks < 8; ++ks) {
        eqf[ks] = loadA(EQb + rowQ * EXP_DIM, EXP_DIM, ks * 32);
        #pragma unroll
        for (int e = 0; e < 16; ++e) eqf[ks].u[e] = f2bf(bf2f(eqf[ks].u[e]) * cb);
    }

    float m[8], l[8];
    v8f o[4];
    #pragma unroll
    for (int r = 0; r < 8; ++r) { m[r] = -1e30f; l[r] = 0.0f; }
    #pragma unroll
    for (int jn = 0; jn < 4; ++jn)
        #pragma unroll
        for (int r = 0; r < 8; ++r) o[jn][r] = 0.0f;

    u16* pw = &Pw[wave * 16 * 80];

    for (int kv0 = 0; kv0 < SEQ; kv0 += 64) {
        __syncthreads();
        {   // stage V tile transposed: Vt[d][kv]
            const int vr = tid >> 2;            // kv row 0..63
            const int vc = (tid & 3) * 16;      // d segment
            const u16* vp = Vb + ((size_t)b * SEQ + kv0 + vr) * HIDDEN
                               + h * HEAD_DIM + vc;
            u16 tmp[16];
            *(uint4*)&tmp[0] = *(const uint4*)vp;
            *(uint4*)&tmp[8] = *(const uint4*)(vp + 8);
            #pragma unroll
            for (int j = 0; j < 16; ++j)
                Vt[(vc + j) * 80 + vr] = tmp[j];
        }
        __syncthreads();

        // scores: QKᵀ and EQ·EKᵀ into one accumulator chain
        v8f sc[4];
        #pragma unroll
        for (int jn = 0; jn < 4; ++jn) {
            #pragma unroll
            for (int r = 0; r < 8; ++r) sc[jn][r] = 0.0f;
            const u16* kp  = Kb  + ((size_t)b * SEQ + kv0 + jn * 16) * HIDDEN
                                 + h * HEAD_DIM;
            #pragma unroll
            for (int ks = 0; ks < 2; ++ks) {
                FragU kf = loadB(kp, HIDDEN, ks * 32);
                sc[jn] = wmma_bf16(qf[ks], kf, sc[jn]);
            }
            const u16* ekp = EKb + ((size_t)b * SEQ + kv0 + jn * 16) * EXP_DIM;
            #pragma unroll
            for (int ks = 0; ks < 8; ++ks) {
                FragU ekf = loadB(ekp, EXP_DIM, ks * 32);
                sc[jn] = wmma_bf16(eqf[ks], ekf, sc[jn]);
            }
        }

        // online softmax (rows live across 16-lane halves)
        #pragma unroll
        for (int r = 0; r < 8; ++r) {
            float rm = fmaxf(fmaxf(sc[0][r], sc[1][r]), fmaxf(sc[2][r], sc[3][r]));
            #pragma unroll
            for (int off = 1; off < 16; off <<= 1)
                rm = fmaxf(rm, __shfl_xor(rm, off, 32));
            const float mnew  = fmaxf(m[r], rm);
            const float alpha = __expf(m[r] - mnew);
            float rs = 0.0f;
            #pragma unroll
            for (int jn = 0; jn < 4; ++jn) {
                float p = __expf(sc[jn][r] - mnew);
                sc[jn][r] = p;
                rs += p;
            }
            #pragma unroll
            for (int off = 1; off < 16; off <<= 1)
                rs += __shfl_xor(rs, off, 32);
            l[r] = l[r] * alpha + rs;
            m[r] = mnew;
            #pragma unroll
            for (int jn = 0; jn < 4; ++jn) o[jn][r] *= alpha;
        }

        // P (D layout) -> per-wave LDS -> A layout; O += P @ V
        #pragma unroll
        for (int jn = 0; jn < 4; ++jn)
            #pragma unroll
            for (int r = 0; r < 8; ++r)
                pw[(hi ? r + 8 : r) * 80 + jn * 16 + l16] = f2bf(sc[jn][r]);

        #pragma unroll
        for (int ks = 0; ks < 2; ++ks) {
            FragU pf = loadA(pw, 80, ks * 32);
            #pragma unroll
            for (int jn = 0; jn < 4; ++jn) {
                FragU vf = loadB(&Vt[(jn * 16) * 80], 80, ks * 32);
                o[jn] = wmma_bf16(pf, vf, o[jn]);
            }
        }
    }

    // normalize and store ctx (bf16)
    #pragma unroll
    for (int r = 0; r < 8; ++r) {
        const float inv = 1.0f / l[r];
        const int qrow = qw + r + (hi ? 8 : 0);
        #pragma unroll
        for (int jn = 0; jn < 4; ++jn)
            Ctxb[((size_t)b * SEQ + qrow) * HIDDEN + h * HEAD_DIM + jn * 16 + l16] =
                f2bf(o[jn][r] * inv);
    }
}

// ---------------------------------------------------------------------------
extern "C" void kernel_launch(void* const* d_in, const int* in_sizes, int n_in,
                              void* d_out, int out_size, void* d_ws, size_t ws_size,
                              hipStream_t stream) {
    const float* x    = (const float*)d_in[0];
    const float* conf = (const float*)d_in[1];
    const float* Wq   = (const float*)d_in[2];
    const float* bq   = (const float*)d_in[3];
    const float* Wk   = (const float*)d_in[4];
    const float* bk   = (const float*)d_in[5];
    const float* Wv   = (const float*)d_in[6];
    const float* bv   = (const float*)d_in[7];
    const float* Weq  = (const float*)d_in[8];
    const float* beq  = (const float*)d_in[9];
    const float* Wek  = (const float*)d_in[10];
    const float* bek  = (const float*)d_in[11];
    const float* Wo   = (const float*)d_in[12];
    const float* bo   = (const float*)d_in[13];
    float* out = (float*)d_out;

    const size_t BS = (size_t)BATCH * SEQ;   // 8192 rows

    u16* xb   = (u16*)d_ws;                  // bf16 x; later reused as ctx
    u16* Qb   = xb   + BS * HIDDEN;
    u16* Kb   = Qb   + BS * HIDDEN;
    u16* Vb   = Kb   + BS * HIDDEN;
    u16* EQb  = Vb   + BS * HIDDEN;
    u16* EKb  = EQb  + BS * EXP_DIM;
    u16* Wqt  = EKb  + BS * EXP_DIM;         // pre-transposed bf16 weights [N][K]
    u16* Wkt  = Wqt  + (size_t)HIDDEN * HIDDEN;
    u16* Wvt  = Wkt  + (size_t)HIDDEN * HIDDEN;
    u16* Wot  = Wvt  + (size_t)HIDDEN * HIDDEN;
    u16* Weqt = Wot  + (size_t)HIDDEN * HIDDEN;
    u16* Wekt = Weqt + (size_t)HIDDEN * EXP_DIM;
    u16* Ctxb = xb;                          // alias: x dead after projections

    dim3 blk(256);

    // bf16 conversion + weight pre-transpose (bandwidth-trivial)
    convert_bf16_kernel<<<dim3((unsigned)(BS * HIDDEN / 1024)), blk, 0, stream>>>(
        x, xb, (int)(BS * HIDDEN));
    dim3 gT(HIDDEN / 32, HIDDEN / 32);       // (32, 32)
    dim3 gTe(HIDDEN / 32, EXP_DIM / 32);     // (32, 8)
    transpose_bf16_kernel<<<gT,  blk, 0, stream>>>(Wq,  Wqt,  HIDDEN, HIDDEN);
    transpose_bf16_kernel<<<gT,  blk, 0, stream>>>(Wk,  Wkt,  HIDDEN, HIDDEN);
    transpose_bf16_kernel<<<gT,  blk, 0, stream>>>(Wv,  Wvt,  HIDDEN, HIDDEN);
    transpose_bf16_kernel<<<gT,  blk, 0, stream>>>(Wo,  Wot,  HIDDEN, HIDDEN);
    transpose_bf16_kernel<<<gTe, blk, 0, stream>>>(Weq, Weqt, HIDDEN, EXP_DIM);
    transpose_bf16_kernel<<<gTe, blk, 0, stream>>>(Wek, Wekt, HIDDEN, EXP_DIM);

    // projections (async-DMA double-buffered WMMA GEMMs)
    dim3 gP(HIDDEN / 128, (unsigned)(BS / 128));     // (8, 64)
    dim3 gE(EXP_DIM / 128, (unsigned)(BS / 128));    // (2, 64)
    gemm_async_kernel<u16><<<gP, blk, 0, stream>>>(xb, Wqt,  bq,  Qb,
        (int)BS, HIDDEN, HIDDEN);
    gemm_async_kernel<u16><<<gP, blk, 0, stream>>>(xb, Wkt,  bk,  Kb,
        (int)BS, HIDDEN, HIDDEN);
    gemm_async_kernel<u16><<<gP, blk, 0, stream>>>(xb, Wvt,  bv,  Vb,
        (int)BS, HIDDEN, HIDDEN);
    gemm_async_kernel<u16><<<gE, blk, 0, stream>>>(xb, Weqt, beq, EQb,
        (int)BS, EXP_DIM, HIDDEN);
    gemm_async_kernel<u16><<<gE, blk, 0, stream>>>(xb, Wekt, bek, EKb,
        (int)BS, EXP_DIM, HIDDEN);

    // attention (flash, fused experience bias)
    dim3 gF(SEQ / 128, NUM_HEADS, BATCH);            // (16, 16, 4)
    flash_attn_kernel<<<gF, blk, 0, stream>>>(Qb, Kb, Vb, EQb, EKb, conf, Ctxb);

    // output projection -> fp32
    gemm_async_kernel<float><<<gP, blk, 0, stream>>>(Ctxb, Wot, bo, out,
        (int)BS, HIDDEN, HIDDEN);
}
